// BinarizeLinear_test_19516331393233
// MI455X (gfx1250) — 8-bit matrix operand order repaired and checked by reading, compiled, not run on hardware
//
#include <hip/hip_runtime.h>


typedef __attribute__((ext_vector_type(16))) int   v16i;
typedef __attribute__((ext_vector_type(8)))  float v8f;

#define FP8_P1 0x38u
#define FP8_M1 0xB8u

__global__ __launch_bounds__(256) void pack_x_fp8(
    const float* __restrict__ x, unsigned int* __restrict__ apk,
    int ktiles, long total, int din)
{
  long t = (long)blockIdx.x * blockDim.x + threadIdx.x;
  if (t >= total) return;
  long blk    = t >> 9;
  int  within = (int)(t & 511);
  int  lane   = within >> 4;
  int  d      = within & 15;
  int  tm     = (int)(blk / ktiles);
  int  kt     = (int)(blk - (long)tm * ktiles);
  int  m      = lane & 15;
  int  half   = lane >> 4;
  int  klocal = ((d >> 1) << 4) + (half << 3) + ((d & 1) << 2);
  long row    = (long)tm * 16 + m;
  long k      = (long)kt * 128 + klocal;
  const float4 f = *(const float4*)(x + row * din + k);
  unsigned u =  (f.x > 0.f ? FP8_P1 : FP8_M1)
             | ((f.y > 0.f ? FP8_P1 : FP8_M1) << 8)
             | ((f.z > 0.f ? FP8_P1 : FP8_M1) << 16)
             | ((f.w > 0.f ? FP8_P1 : FP8_M1) << 24);
  apk[t] = u;
}

__global__ __launch_bounds__(256) void pack_w_fp8(
    const float* __restrict__ wgt, unsigned int* __restrict__ wpk,
    int ktiles, long total, int din)
{
  long t = (long)blockIdx.x * blockDim.x + threadIdx.x;
  if (t >= total) return;
  long blk    = t >> 9;
  int  within = (int)(t & 511);
  int  lane   = within >> 4;
  int  d      = within & 15;
  int  tn     = (int)(blk / ktiles);
  int  kt     = (int)(blk - (long)tn * ktiles);
  int  n      = lane & 15;
  int  half   = lane >> 4;
  int  klocal = ((d >> 1) << 4) + (half << 3) + ((d & 1) << 2);
  long row    = (long)tn * 16 + n;
  long k      = (long)kt * 128 + klocal;
  const float4 f = *(const float4*)(wgt + row * din + k);
  unsigned u =  (f.x > 0.f ? FP8_P1 : FP8_M1)
             | ((f.y > 0.f ? FP8_P1 : FP8_M1) << 8)
             | ((f.z > 0.f ? FP8_P1 : FP8_M1) << 16)
             | ((f.w > 0.f ? FP8_P1 : FP8_M1) << 24);
  wpk[t] = u;
}

__global__ __launch_bounds__(256) void bgemm_fp8_wmma(
    const unsigned char* __restrict__ apk,
    const unsigned char* __restrict__ wpk,
    const float* __restrict__ bias,
    float* __restrict__ out,
    int nblk_n, int ktiles, int dout)
{
  const int lane   = threadIdx.x & 31;
  const int waveid = threadIdx.x >> 5;
  const int bm     = blockIdx.x / nblk_n;
  const int bn     = blockIdx.x - bm * nblk_n;
  const int wml    = waveid & 3;
  const int wnl    = waveid >> 2;

  const int tm0 = bm * 8 + wml * 2;
  const int tn0 = bn * 8 + wnl * 4;

  const long laneoff = (long)lane * 64;
  const unsigned char* a0p = apk + ((long)(tm0 + 0) * ktiles) * 2048 + laneoff;
  const unsigned char* a1p = apk + ((long)(tm0 + 1) * ktiles) * 2048 + laneoff;
  const unsigned char* b0p = wpk + ((long)(tn0 + 0) * ktiles) * 2048 + laneoff;
  const unsigned char* b1p = wpk + ((long)(tn0 + 1) * ktiles) * 2048 + laneoff;
  const unsigned char* b2p = wpk + ((long)(tn0 + 2) * ktiles) * 2048 + laneoff;
  const unsigned char* b3p = wpk + ((long)(tn0 + 3) * ktiles) * 2048 + laneoff;

  v8f acc[2][4];
#pragma unroll
  for (int i = 0; i < 2; ++i)
#pragma unroll
    for (int j = 0; j < 4; ++j)
      acc[i][j] = (v8f){0.f, 0.f, 0.f, 0.f, 0.f, 0.f, 0.f, 0.f};

  for (int kt = 0; kt < ktiles; ++kt) {
    const long ko = (long)kt * 2048;
    v16i a0 = *(const v16i*)(a0p + ko);
    v16i a1 = *(const v16i*)(a1p + ko);
    v16i b0 = *(const v16i*)(b0p + ko);
    v16i b1 = *(const v16i*)(b1p + ko);
    v16i b2 = *(const v16i*)(b2p + ko);
    v16i b3 = *(const v16i*)(b3p + ko);

    acc[0][0] = __builtin_amdgcn_wmma_f32_16x16x128_fp8_fp8(a0, b0, (short)0, acc[0][0], false, false);
    acc[0][1] = __builtin_amdgcn_wmma_f32_16x16x128_fp8_fp8(a0, b1, (short)0, acc[0][1], false, false);
    acc[0][2] = __builtin_amdgcn_wmma_f32_16x16x128_fp8_fp8(a0, b2, (short)0, acc[0][2], false, false);
    acc[0][3] = __builtin_amdgcn_wmma_f32_16x16x128_fp8_fp8(a0, b3, (short)0, acc[0][3], false, false);
    acc[1][0] = __builtin_amdgcn_wmma_f32_16x16x128_fp8_fp8(a1, b0, (short)0, acc[1][0], false, false);
    acc[1][1] = __builtin_amdgcn_wmma_f32_16x16x128_fp8_fp8(a1, b1, (short)0, acc[1][1], false, false);
    acc[1][2] = __builtin_amdgcn_wmma_f32_16x16x128_fp8_fp8(a1, b2, (short)0, acc[1][2], false, false);
    acc[1][3] = __builtin_amdgcn_wmma_f32_16x16x128_fp8_fp8(a1, b3, (short)0, acc[1][3], false, false);
  }

  const int ncol = lane & 15;
  const int half = lane >> 4;
#pragma unroll
  for (int j = 0; j < 4; ++j) {
    const int   col = (tn0 + j) * 16 + ncol;
    const float bv  = bias[col];
#pragma unroll
    for (int i = 0; i < 2; ++i) {
      const long rowbase = (long)(tm0 + i) * 16 + half * 8;
      float* p = out + rowbase * dout + col;
      v8f c = acc[i][j];
#pragma unroll
      for (int r = 0; r < 8; ++r)
        __builtin_nontemporal_store(c[r] + bv, p + (long)r * dout);
    }
  }
}

extern "C" void kernel_launch(void* const* d_in, const int* in_sizes, int n_in,
                              void* d_out, int out_size, void* d_ws, size_t ws_size,
                              hipStream_t stream) {
  const float* x    = (const float*)d_in[0];
  const float* wgt  = (const float*)d_in[1];
  const float* bias = (const float*)d_in[2];
  float*       out  = (float*)d_out;

  const int  dout = in_sizes[2];
  const int  din  = (int)((long)in_sizes[1] / dout);
  const int  n    = (int)((long)in_sizes[0] / din);
  const int  ktiles = din / 128;

  unsigned char* apk = (unsigned char*)d_ws;
  unsigned char* wpk = apk + (size_t)n * din;

  const long a_dwords = (long)n * din / 4;
  const long w_dwords = (long)dout * din / 4;
  pack_x_fp8<<<(int)((a_dwords + 255) / 256), 256, 0, stream>>>(
      x, (unsigned int*)apk, ktiles, a_dwords, din);
  pack_w_fp8<<<(int)((w_dwords + 255) / 256), 256, 0, stream>>>(
      wgt, (unsigned int*)wpk, ktiles, w_dwords, din);

  const int nblk_m = n / 128;
  const int nblk_n = dout / 128;
  bgemm_fp8_wmma<<<nblk_m * nblk_n, 256, 0, stream>>>(
      apk, wpk, bias, out, nblk_n, ktiles, dout);
}
